// LowRankSparseAttention_54288386621725
// MI455X (gfx1250) — compile-verified, matrix operand order checked by reading, not run on hardware
//
#include <hip/hip_runtime.h>
#include <hip/hip_bf16.h>

typedef __attribute__((ext_vector_type(16))) __bf16 v16bf;
typedef __attribute__((ext_vector_type(8)))  __bf16 v8bf;
typedef __attribute__((ext_vector_type(8)))  float  v8f;

#define DEVI __device__ __forceinline__

namespace {

constexpr int B_   = 2;
constexpr int SEQ_ = 1024;
constexpr int DM_  = 1024;
constexpr int HQK_ = 32;    // N_QK heads
constexpr int DQK_ = 32;    // head dim
constexpr int NOV_ = 2048;  // N_OV (value width after regroup: 32 heads x 64)
constexpr int VKV_ = 16;
constexpr int KVP_ = 1056;  // kv length 1040 padded to multiple of 32 (pad always masked)
constexpr int NQKV_ = 4096; // fused projection width: 1024 Q | 1024 K | 2048 V
constexpr float INV_SCALE_ = 0.17677669529663687f; // 1/sqrt(32)

DEVI __bf16 f2bf(float f) {
  unsigned u = __float_as_uint(f);
  unsigned short s = (unsigned short)((u + 0x7FFFu + ((u >> 16) & 1u)) >> 16); // RNE
  union { unsigned short s; __bf16 b; } cv; cv.s = s;
  return cv.b;
}

DEVI v8f v8f_zero() {
  v8f z;
#pragma unroll
  for (int i = 0; i < 8; ++i) z[i] = 0.0f;
  return z;
}

// Load a 16x32 bf16 WMMA fragment from a row-major [16][ld] tile (global or LDS).
// ISA 16-bit A layout: lane<16 -> row=lane,   elems 0..7 = K 0..7,  elems 8..15 = K 16..23
//                      lane>=16 -> row=lane-16, elems 0..7 = K 8..15, elems 8..15 = K 24..31
DEVI v16bf load_frag(const __bf16* base, int ld, int lane) {
  const int l = lane & 15, hi = lane >> 4;
  const __bf16* p = base + (size_t)l * ld + hi * 8;
  union { v16bf v; v8bf h[2]; } f;
  f.h[0] = *(const v8bf*)(p);
  f.h[1] = *(const v8bf*)(p + 16);
  return f.v;
}

DEVI float rope_val(float x, float xp, int d, int pos) {
  // reference: rot = x*cos + flip*sin ; flip[d] = d<16 ? -x[d+16] : x[d-16]
  float flip = (d < 16) ? -xp : xp;
  float ang = (float)pos * __powf(10000.0f, -(float)(d & 15) * (1.0f / 16.0f));
  return x * __cosf(ang) + flip * __sinf(ang);
}

// Async copy of one 16B chunk: global -> LDS, tracked by ASYNCcnt.
DEVI void async_b128(unsigned lds_byte_addr, const void* gaddr) {
  asm volatile("global_load_async_to_lds_b128 %0, %1, off"
               :: "v"(lds_byte_addr), "v"((unsigned long long)gaddr)
               : "memory");
}

DEVI unsigned lds_addr(const void* p) {
  // generic shared pointer = LDS aperture | offset; low 32 bits are the LDS byte address
  return (unsigned)(unsigned long long)p;
}

} // namespace

// ---------------- pack kernels ----------------

__global__ __launch_bounds__(256) void k_pack_resid(const float* __restrict__ x,
                                                    __bf16* __restrict__ y, int n) {
  int i = blockIdx.x * 256 + threadIdx.x;
  if (i < n) y[i] = f2bf(x[i]);
}

// wqkv_t[n][m], n in [0,4096): cols of fused QKV weight, stored N-major for B-fragment loads
__global__ __launch_bounds__(256) void k_pack_wqkv(const float* __restrict__ WQ,
                                                   const float* __restrict__ WK,
                                                   const float* __restrict__ WV,
                                                   __bf16* __restrict__ wt) {
  int i = blockIdx.x * 256 + threadIdx.x;
  if (i >= NQKV_ * DM_) return;
  int n = i >> 10;
  int m = i & 1023;
  float v;
  if (n < 1024) {
    int h = n >> 5, d = n & 31;
    v = WQ[((size_t)h * DM_ + m) * DQK_ + d];
  } else if (n < 2048) {
    int h = (n - 1024) >> 5, d = n & 31;
    v = WK[((size_t)h * DM_ + m) * DQK_ + d];
  } else {
    v = WV[(size_t)(n - 2048) * DM_ + m]; // W_V[h2][m][0]
  }
  wt[(size_t)n * DM_ + m] = f2bf(v);
}

// wo_t[n][k] = W_O[k][0][n], N-major
__global__ __launch_bounds__(256) void k_pack_wo(const float* __restrict__ WO,
                                                 __bf16* __restrict__ wt) {
  int i = blockIdx.x * 256 + threadIdx.x;
  if (i >= DM_ * NOV_) return;
  int n = i >> 11;
  int k = i & 2047;
  wt[(size_t)n * NOV_ + k] = f2bf(WO[(size_t)k * DM_ + n]);
}

// ---------------- bf16 WMMA GEMM, async double-buffered LDS ----------------
// C[M][N] = A[M][K] * Bt[N][K]^T.  Block tile 128(M) x 64(N): 8 waves x (16x64).
// A/B k-slices staged in LDS via GLOBAL_LOAD_ASYNC_TO_LDS_B128 (ASYNCcnt), ping-pong.

__global__ __launch_bounds__(256) void k_gemm_bf16(const __bf16* __restrict__ A,
                                                   const __bf16* __restrict__ Bt,
                                                   float* __restrict__ C,
                                                   int M, int N, int K) {
  __shared__ __align__(16) __bf16 tileA[2][128][32];
  __shared__ __align__(16) __bf16 tileB[2][64][32];

  const int tid = threadIdx.x;
  const int lane = tid & 31, w = tid >> 5;
  const int l16 = lane & 15, hi = lane >> 4;

  const int nblk = N >> 6;
  const int m0 = (blockIdx.x / nblk) << 7;
  const int n0 = (blockIdx.x % nblk) << 6;

  // per-thread async chunk assignment (16B chunks)
  // A: 128 rows x 64B; thread t copies row t>>1, byte-half (t&1): two b128s.
  //    -> wave w copies exactly its own 16 rows [16w, 16w+16).
  const int arow = tid >> 1, acol = (tid & 1) * 16;  // element offsets
  // B: 64 rows x 64B; thread t copies row t>>2, chunk t&3: one b128 (shared by all waves).
  const int brow = tid >> 2, bcol = (tid & 3) * 8;

  auto issue = [&](int buf, int k0) {
    const __bf16* ga = A + (size_t)(m0 + arow) * K + k0 + acol;
    const __bf16* gb = Bt + (size_t)(n0 + brow) * K + k0 + bcol;
    async_b128(lds_addr(&tileA[buf][arow][acol]), ga);
    async_b128(lds_addr(&tileA[buf][arow][acol + 8]), ga + 8);
    async_b128(lds_addr(&tileB[buf][brow][bcol]), gb);
  };

  v8f acc[4];
#pragma unroll
  for (int f = 0; f < 4; ++f) acc[f] = v8f_zero();

  issue(0, 0);
  int buf = 0;
  for (int k0 = 0; k0 < K; k0 += 32) {
    if (k0 + 32 < K) {
      issue(buf ^ 1, k0 + 32);                         // prefetch next slice
      asm volatile("s_wait_asynccnt 0x3" ::: "memory"); // current slice (3 oldest) landed
    } else {
      asm volatile("s_wait_asynccnt 0x0" ::: "memory");
    }
    __syncthreads(); // all waves' pieces of this buffer visible

    v16bf a = load_frag(&tileA[buf][w << 4][0], 32, lane);
#pragma unroll
    for (int f = 0; f < 4; ++f) {
      v16bf b = load_frag(&tileB[buf][f << 4][0], 32, lane);
      acc[f] = __builtin_amdgcn_wmma_f32_16x16x32_bf16(false, a, false, b,
                                                       (short)0, acc[f], false, false);
    }
    __syncthreads(); // everyone done reading before this buffer is overwritten
    buf ^= 1;
  }

  // C layout: vgpr r -> row m0 + 16w + r + 8*hi ; col = n0 + f*16 + (lane&15)
#pragma unroll
  for (int f = 0; f < 4; ++f)
#pragma unroll
    for (int r = 0; r < 8; ++r)
      C[(size_t)(m0 + (w << 4) + r + (hi << 3)) * N + n0 + (f << 4) + l16] = acc[f][r];
}

// ---------------- RoPE / layout kernels ----------------

__global__ __launch_bounds__(256) void k_build_q(const float* __restrict__ QKV,
                                                 const float* __restrict__ bQ,
                                                 __bf16* __restrict__ q) {
  int i = blockIdx.x * 256 + threadIdx.x; // over B*SEQ*H*D = 2M
  if (i >= B_ * SEQ_ * HQK_ * DQK_) return;
  int d = i & 31, h = (i >> 5) & 31, s = (i >> 10) & 1023, b = i >> 20;
  int dp = (d < 16) ? d + 16 : d - 16;
  const float* row = QKV + ((size_t)(b << 10) + s) * NQKV_ + h * 32;
  float x  = row[d]  + bQ[h * 32 + d];
  float xp = row[dp] + bQ[h * 32 + dp];
  q[((size_t)(b * HQK_ + h) * SEQ_ + s) * DQK_ + d] = f2bf(rope_val(x, xp, d, s));
}

__global__ __launch_bounds__(256) void k_build_k(const float* __restrict__ QKV,
                                                 const float* __restrict__ bK,
                                                 const float* __restrict__ vk,
                                                 __bf16* __restrict__ kbuf) {
  int i = blockIdx.x * 256 + threadIdx.x; // over B*H*KVP*D
  if (i >= B_ * HQK_ * KVP_ * DQK_) return;
  int d = i & 31;
  int t = i >> 5;
  int kv = t % KVP_;
  int t2 = t / KVP_;
  int h = t2 & 31, b = t2 >> 5;
  float val;
  if (kv < SEQ_) {
    int dp = (d < 16) ? d + 16 : d - 16;
    const float* row = QKV + ((size_t)(b << 10) + kv) * NQKV_ + 1024 + h * 32;
    float x  = row[d]  + bK[h * 32 + d];
    float xp = row[dp] + bK[h * 32 + dp];
    val = rope_val(x, xp, d, kv);
  } else if (kv < SEQ_ + VKV_) {
    val = vk[((size_t)(kv - SEQ_) * HQK_ + h) * DQK_ + d]; // no rotary on virtual keys
  } else {
    val = 0.0f; // padding (always masked)
  }
  kbuf[((size_t)(b * HQK_ + h) * KVP_ + kv) * DQK_ + d] = f2bf(val);
}

// V transposed: vt[b][n][kv], n in [0,2048), so B-fragments load contiguous kv runs
__global__ __launch_bounds__(256) void k_build_vt(const float* __restrict__ QKV,
                                                  const float* __restrict__ bV,
                                                  const float* __restrict__ vv,
                                                  __bf16* __restrict__ vt) {
  int i = blockIdx.x * 256 + threadIdx.x; // over B*NOV*KVP
  if (i >= B_ * NOV_ * KVP_) return;
  int kv = i % KVP_;
  int t = i / KVP_;
  int n = t & 2047, b = t >> 11;
  float val;
  if (kv < SEQ_) {
    val = QKV[((size_t)(b << 10) + kv) * NQKV_ + 2048 + n] + bV[n];
  } else if (kv < SEQ_ + VKV_) {
    val = vv[(size_t)(kv - SEQ_) * NOV_ + n];
  } else {
    val = 0.0f;
  }
  vt[((size_t)b * NOV_ + n) * KVP_ + kv] = f2bf(val);
}

// ---------------- flash-style attention: one wave per (b, h, 16-query tile) ----------------

__global__ __launch_bounds__(256) void k_attn(const __bf16* __restrict__ q,
                                              const __bf16* __restrict__ kbuf,
                                              const __bf16* __restrict__ vt,
                                              __bf16* __restrict__ z) {
  __shared__ float Ssc[8][16][32];
  __shared__ __align__(16) __bf16 Pp[8][16][32];
  __shared__ float mrow[8][16];
  __shared__ float lrow[8][16];
  __shared__ float crow[8][16];

  const int w = threadIdx.x >> 5, lane = threadIdx.x & 31;
  const int task = blockIdx.x * 8 + w;
  if (task >= B_ * HQK_ * (SEQ_ / 16)) return; // wave-uniform
  const int qt = task & 63;
  const int h  = (task >> 6) & 31;
  const int b  = task >> 11;
  const int q0 = qt << 4;
  const int l16 = lane & 15, hi = lane >> 4;

  if (lane < 16) { mrow[w][lane] = -__builtin_inff(); lrow[w][lane] = 0.0f; }

  v8f acc[4];
#pragma unroll
  for (int f = 0; f < 4; ++f) acc[f] = v8f_zero();

  const v16bf aq = load_frag(q + ((size_t)(b * HQK_ + h) * SEQ_ + q0) * DQK_, DQK_, lane);
  const __bf16* kb = kbuf + (size_t)(b * HQK_ + h) * KVP_ * DQK_;
  const __bf16* vb = vt + ((size_t)b * NOV_ + h * 64) * KVP_;

  const int nch = (q0 + 63) >> 5; // covers max allowed key q0+15+16
  for (int c = 0; c < nch; ++c) {
    const int kv0 = c << 5;
    if (c + 1 < nch) { // prefetch next kv chunk (K rows + V columns)
      __builtin_prefetch(kb + (size_t)(kv0 + 32) * DQK_ + (size_t)l16 * DQK_, 0, 0);
      __builtin_prefetch(vb + (size_t)l16 * 4 * KVP_ + kv0 + 32, 0, 0);
    }
    v16bf bk0 = load_frag(kb + (size_t)kv0 * DQK_, DQK_, lane);
    v16bf bk1 = load_frag(kb + (size_t)(kv0 + 16) * DQK_, DQK_, lane);
    v8f zc = v8f_zero();
    v8f s0 = __builtin_amdgcn_wmma_f32_16x16x32_bf16(false, aq, false, bk0, (short)0, zc, false, false);
    v8f s1 = __builtin_amdgcn_wmma_f32_16x16x32_bf16(false, aq, false, bk1, (short)0, zc, false, false);

    // scale + mask (k <= q + VKV), scatter C layout -> row-major LDS tile [16 q][32 kv]
#pragma unroll
    for (int r = 0; r < 8; ++r) {
      int qq = q0 + r + (hi << 3);
      int ki = kv0 + l16;
      float v0 = (ki      <= qq + VKV_) ? s0[r] * INV_SCALE_ : -__builtin_inff();
      float v1 = (ki + 16 <= qq + VKV_) ? s1[r] * INV_SCALE_ : -__builtin_inff();
      Ssc[w][r + (hi << 3)][l16]      = v0;
      Ssc[w][r + (hi << 3)][l16 + 16] = v1;
    }

    // online softmax row stats (one lane per query row); same-wave LDS is in-order
    if (lane < 16) {
      const int row = lane;
      float mx = mrow[w][row];
      float cm = -__builtin_inff();
#pragma unroll
      for (int j = 0; j < 32; ++j) cm = fmaxf(cm, Ssc[w][row][j]);
      float nm = fmaxf(mx, cm);          // finite: key 0 always unmasked
      float corr = __expf(mx - nm);      // first chunk: exp(-inf) = 0, acc is 0
      float sum = 0.0f;
#pragma unroll
      for (int j = 0; j < 32; ++j) {
        float e = __expf(Ssc[w][row][j] - nm); // masked -> exp(-inf) = 0
        Pp[w][row][j] = f2bf(e);
        sum += e;
      }
      lrow[w][row] = lrow[w][row] * corr + sum;
      mrow[w][row] = nm;
      crow[w][row] = corr;
    }

    // rescale accumulators by per-row correction
#pragma unroll
    for (int r = 0; r < 8; ++r) {
      float cc = crow[w][r + (hi << 3)];
#pragma unroll
      for (int f = 0; f < 4; ++f) acc[f][r] *= cc;
    }

    // P (16x32) from LDS as A fragment; V chunk (32 x 64) as 4 B fragments
    v16bf ap = load_frag(&Pp[w][0][0], 32, lane);
#pragma unroll
    for (int f = 0; f < 4; ++f) {
      v16bf bv = load_frag(vb + (size_t)(f << 4) * KVP_ + kv0, KVP_, lane);
      acc[f] = __builtin_amdgcn_wmma_f32_16x16x32_bf16(false, ap, false, bv, (short)0, acc[f], false, false);
    }
  }

  if (lane < 16) {
    float l = lrow[w][lane];
    crow[w][lane] = (l > 0.0f) ? 1.0f / l : 0.0f;
  }
#pragma unroll
  for (int r = 0; r < 8; ++r) {
    float inv = crow[w][r + (hi << 3)];
    size_t grow = (size_t)(b << 10) + q0 + r + (hi << 3);
#pragma unroll
    for (int f = 0; f < 4; ++f)
      z[grow * NOV_ + h * 64 + (f << 4) + l16] = f2bf(acc[f][r] * inv);
  }
}

// ---------------- host launcher ----------------

extern "C" void kernel_launch(void* const* d_in, const int* in_sizes, int n_in,
                              void* d_out, int out_size, void* d_ws, size_t ws_size,
                              hipStream_t stream) {
  const float* resid = (const float*)d_in[0];
  const float* W_Q   = (const float*)d_in[1];
  const float* W_K   = (const float*)d_in[2];
  const float* W_V   = (const float*)d_in[3];
  const float* W_O   = (const float*)d_in[4];
  const float* b_Q   = (const float*)d_in[5];
  const float* b_K   = (const float*)d_in[6];
  const float* b_V   = (const float*)d_in[7];
  const float* vk    = (const float*)d_in[8];
  const float* vv    = (const float*)d_in[9];
  float* out = (float*)d_out;
  (void)in_sizes; (void)n_in; (void)out_size; (void)ws_size;

  char* ws = (char*)d_ws;
  size_t off = 0;
  auto take = [&](size_t bytes) -> char* {
    char* p = ws + off;
    off = (off + bytes + 255) & ~(size_t)255;
    return p;
  };
  __bf16* resid_bf = (__bf16*)take((size_t)2048 * 1024 * 2);        // A of QKV gemm
  __bf16* wqkv_t   = (__bf16*)take((size_t)4096 * 1024 * 2);        // Bt of QKV gemm
  __bf16* wo_t     = (__bf16*)take((size_t)1024 * 2048 * 2);        // Bt of out gemm
  float*  qkv      = (float*) take((size_t)2048 * 4096 * 4);        // fused projection
  __bf16* qbuf     = (__bf16*)take((size_t)2 * 32 * 1024 * 32 * 2); // [b][h][s][d]
  __bf16* kbuf     = (__bf16*)take((size_t)2 * 32 * 1056 * 32 * 2); // [b][h][kvp][d]
  __bf16* vtbuf    = (__bf16*)take((size_t)2 * 2048 * 1056 * 2);    // [b][n][kvp]
  __bf16* zbuf     = (__bf16*)take((size_t)2048 * 2048 * 2);        // A of out gemm

  k_pack_resid<<<(2048 * 1024 + 255) / 256, 256, 0, stream>>>(resid, resid_bf, 2048 * 1024);
  k_pack_wqkv<<<(4096 * 1024 + 255) / 256, 256, 0, stream>>>(W_Q, W_K, W_V, wqkv_t);
  k_pack_wo<<<(1024 * 2048 + 255) / 256, 256, 0, stream>>>(W_O, wo_t);

  // fused QKV projection: [2048 x 1024] x [1024 x 4096]; block tile 128x64
  k_gemm_bf16<<<(2048 / 128) * (4096 / 64), 256, 0, stream>>>(resid_bf, wqkv_t, qkv,
                                                              2048, 4096, 1024);

  k_build_q<<<(2 * 1024 * 32 * 32 + 255) / 256, 256, 0, stream>>>(qkv, b_Q, qbuf);
  k_build_k<<<(2 * 32 * 1056 * 32 + 255) / 256, 256, 0, stream>>>(qkv, b_K, vk, kbuf);
  k_build_vt<<<(2 * 2048 * 1056 + 255) / 256, 256, 0, stream>>>(qkv, b_V, vv, vtbuf);

  // attention: 2*32*64 = 4096 wave-tasks, 8 waves per block
  k_attn<<<4096 / 8, 256, 0, stream>>>(qbuf, kbuf, vtbuf, zbuf);

  // output projection: [2048 x 2048] x [2048 x 1024] -> d_out (f32)
  k_gemm_bf16<<<(2048 / 128) * (1024 / 64), 256, 0, stream>>>(zbuf, wo_t, out,
                                                              2048, 1024, 2048);
}
